// MotionEncoder_45758581571933
// MI455X (gfx1250) — matrix operand order repaired and checked by reading, not compiled, not run on hardware
//
#include <hip/hip_runtime.h>

typedef __attribute__((ext_vector_type(16))) _Float16 v16h;
typedef __attribute__((ext_vector_type(8)))  float    v8f;

#define NA_ 40000
#define NL_ 30000
#define TA_ 20
#define TL_ 10
#define EAA_ 640000
#define EAL_ 1200000
#define BN_INV_ 0.99999500003749973f

__device__ __forceinline__ float sigmf_(float x){ return 1.0f/(1.0f+__expf(-x)); }
__device__ __forceinline__ float tanhf_(float x){ return 2.0f*sigmf_(2.0f*x)-1.0f; }
__device__ __forceinline__ float lrelu02_(float x){ return x>0.0f ? x : 0.2f*x; }

__device__ __forceinline__ void atomicMaxF_(float* a, float v){
  if (v >= 0.0f) atomicMax((int*)a, __float_as_int(v));
  else           atomicMin((unsigned int*)a, (unsigned int)__float_as_int(v));
}

// ---------------- weight transpose + f16 convert: out[n*K+k] = W[k*N+n] -------
__global__ void transconv_f16_kernel(const float* __restrict__ W, _Float16* __restrict__ out,
                                     int K, int N){
  int i = blockIdx.x*blockDim.x + threadIdx.x;
  if (i < N*K){ int n = i / K, k = i - n*K; out[(size_t)n*K + k] = (_Float16)W[(size_t)k*N + n]; }
}

// ---------------- LSTM via v_wmma_f32_16x16x32_f16 ---------------------------
// wave handles 16 rows, full 4H=256 gate width; h state kept f32 in C-layout regs.
template<int D, bool MEAN>
__global__ __launch_bounds__(256, 1) void lstm_wmma_kernel(
    const float* __restrict__ x, const _Float16* __restrict__ WhhT,
    const float* __restrict__ Wih, const float* __restrict__ bih, const float* __restrict__ bhh,
    float* __restrict__ h_out, float* __restrict__ mean_out, int N, int T)
{
  __shared__ __align__(16) _Float16 sWhh[256*72];   // [n][k] padded rows (conflict-free b128)
  __shared__ float sBias[256];
  __shared__ float sWih[D*256];
  __shared__ __align__(16) _Float16 sH[8][16*72];   // per-wave h (f16, [m][k] padded)
  __shared__ float sX[8][16*D];

  const int tid = threadIdx.x;
  for (int i = tid; i < 256*64; i += 256){ int n = i >> 6, k = i & 63; sWhh[n*72+k] = WhhT[i]; }
  if (tid < 256) sBias[tid] = bih[tid] + bhh[tid];
  for (int i = tid; i < D*256; i += 256) sWih[i] = Wih[i];
  __syncthreads();

  const int wave = tid >> 5, lane = tid & 31;
  const int l16 = lane & 15, hlf = lane >> 4;
  const int rowbase = (blockIdx.x*8 + wave)*16;
  const bool fullTile = (rowbase + 16) <= N;

  v8f hC[4], cS[4], mS[4];
  #pragma unroll
  for (int t4=0;t4<4;++t4){
    #pragma unroll
    for (int r=0;r<8;++r){ hC[t4][r]=0.0f; cS[t4][r]=0.0f; mS[t4][r]=0.0f; }
  }

  for (int t=0; t<T; ++t){
    if (lane < 16){
      int rr = rowbase + lane;
      const float* xp = x + (size_t)rr*(T*D) + (size_t)t*D;
      if (fullTile){
        #pragma unroll
        for (int d=0; d<D; ++d) sX[wave][lane*D+d] = xp[d];
        if (t + 1 < T) __builtin_prefetch(xp + D, 0, 0);   // next step -> global_prefetch
      } else {
        #pragma unroll
        for (int d=0; d<D; ++d) sX[wave][lane*D+d] = (rr < N) ? xp[d] : 0.0f;
      }
    }
    // C-layout f32 h  ->  A-layout f16 staging
    #pragma unroll
    for (int tt=0; tt<4; ++tt)
      #pragma unroll
      for (int r=0; r<8; ++r)
        sH[wave][(r+8*hlf)*72 + (l16+16*tt)] = (_Float16)hC[tt][r];
    __syncthreads();

    union AV { v16h v; int4 q[2]; } A[2];
    #pragma unroll
    for (int kt=0; kt<2; ++kt){
      const _Float16* p = &sH[wave][l16*72 + kt*32 + hlf*8];
      A[kt].q[0] = *(const int4*)p;           // K = kt*32 + hlf*8 + 0..7
      A[kt].q[1] = *(const int4*)(p + 16);    // K = kt*32 + 16 + hlf*8 + 0..7
    }

    // acc = bias + x@Wih (tiny K=D input projection on VALU, co-executes with WMMA)
    v8f acc[16];
    #pragma unroll
    for (int nt=0; nt<16; ++nt){
      int n = nt*16 + l16;
      #pragma unroll
      for (int r=0; r<8; ++r){
        float s = sBias[n];
        #pragma unroll
        for (int d=0; d<D; ++d) s += sX[wave][(r+8*hlf)*D+d]*sWih[d*256+n];
        acc[nt][r] = s;
      }
    }
    // acc += h @ Whh : 32 WMMAs
    #pragma unroll
    for (int nt=0; nt<16; ++nt){
      #pragma unroll
      for (int kt=0; kt<2; ++kt){
        union AV b;
        const _Float16* bp = &sWhh[(nt*16+l16)*72 + kt*32 + hlf*8];
        b.q[0] = *(const int4*)bp;
        b.q[1] = *(const int4*)(bp+16);
        acc[nt] = __builtin_amdgcn_wmma_f32_16x16x32_f16(
            false, A[kt].v, false, b.v, (short)0, acc[nt], false, false);
      }
    }
    // gates: cols [0,64)=i [64,128)=f [128,192)=g [192,256)=o -> tiles tt,tt+4,tt+8,tt+12
    #pragma unroll
    for (int tt=0; tt<4; ++tt)
      #pragma unroll
      for (int r=0; r<8; ++r){
        float ig = sigmf_(acc[tt   ][r]);
        float fg = sigmf_(acc[tt+ 4][r]);
        float gg = tanhf_(acc[tt+ 8][r]);
        float og = sigmf_(acc[tt+12][r]);
        float cn = fg*cS[tt][r] + ig*gg;
        cS[tt][r] = cn;
        float hn = og*tanhf_(cn);
        hC[tt][r] = hn;
        if (MEAN) mS[tt][r] += hn;
      }
  }

  if (h_out){
    if (fullTile){
      #pragma unroll
      for (int tt=0; tt<4; ++tt)
        #pragma unroll
        for (int r=0; r<8; ++r)
          h_out[(size_t)(rowbase + r + 8*hlf)*64 + tt*16 + l16] = hC[tt][r];
    } else {
      #pragma unroll
      for (int tt=0; tt<4; ++tt)
        #pragma unroll
        for (int r=0; r<8; ++r){
          int m = rowbase + r + 8*hlf;
          if (m < N) h_out[(size_t)m*64 + tt*16 + l16] = hC[tt][r];
        }
    }
  }
  if (MEAN && mean_out){
    float inv = 1.0f/(float)T;
    if (fullTile){
      #pragma unroll
      for (int tt=0; tt<4; ++tt)
        #pragma unroll
        for (int r=0; r<8; ++r)
          mean_out[(size_t)(rowbase + r + 8*hlf)*64 + tt*16 + l16] = mS[tt][r]*inv;
    } else {
      #pragma unroll
      for (int tt=0; tt<4; ++tt)
        #pragma unroll
        for (int r=0; r<8; ++r){
          int m = rowbase + r + 8*hlf;
          if (m < N) mean_out[(size_t)m*64 + tt*16 + l16] = mS[tt][r]*inv;
        }
    }
  }
}

// ---------------- [N,64]@[64,64] WMMA GEMM + fused attention-logit reduction --
__device__ __forceinline__ void att_reduce_store_(const v8f* acc, const float* __restrict__ att,
                                                  float* __restrict__ aout, int heads,
                                                  int base, int hlf, int l16, int N){
  float red[4][8];
  #pragma unroll
  for (int nt=0; nt<4; ++nt){
    float av = att[nt*16 + l16];
    #pragma unroll
    for (int r=0; r<8; ++r){
      float v = acc[nt][r]*av;
      #pragma unroll
      for (int mk=1; mk<16; mk<<=1) v += __shfl_xor(v, mk, 32);
      red[nt][r] = v;
    }
  }
  if (l16 == 0){
    int th = 4/heads;   // n-tiles per head
    for (int h=0; h<heads; ++h)
      #pragma unroll
      for (int r=0; r<8; ++r){
        float s = 0.0f;
        for (int tt=h*th; tt<h*th+th; ++tt) s += red[tt][r];
        int m = base + r + 8*hlf;
        if (m < N) aout[(size_t)m*heads + h] = s;
      }
  }
}

__global__ __launch_bounds__(256) void gemm64_wmma_kernel(
    const float* __restrict__ X, const _Float16* __restrict__ WT, const float* __restrict__ bias,
    float* __restrict__ Y, const float* __restrict__ att1, float* __restrict__ a1,
    const float* __restrict__ att2, float* __restrict__ a2, int heads, int N)
{
  const int tid = threadIdx.x, wave = tid>>5, lane = tid&31, l16 = lane&15, hlf = lane>>4;
  const int base = (blockIdx.x*8 + wave)*16;
  if (base >= N) return;                       // uniform per wave; no barriers below
  const int row = base + l16;
  const bool fullTile = (base + 16) <= N;

  // preload all 8 loop-invariant B fragments (one load clause, no per-WMMA waits)
  union AV { v16h v; int4 q[2]; _Float16 e[16]; };
  AV B[8];
  #pragma unroll
  for (int nt=0; nt<4; ++nt)
    #pragma unroll
    for (int kt=0; kt<2; ++kt){
      const _Float16* bp = WT + (nt*16+l16)*64 + kt*32 + hlf*8;
      B[nt*2+kt].q[0] = *(const int4*)bp;
      B[nt*2+kt].q[1] = *(const int4*)(bp+16);
    }

  AV A[2];
  #pragma unroll
  for (int kt=0; kt<2; ++kt){ A[kt].q[0] = make_int4(0,0,0,0); A[kt].q[1] = make_int4(0,0,0,0); }
  if (fullTile || row < N){
    const float* xr = X + (size_t)row*64;
    #pragma unroll
    for (int kt=0; kt<2; ++kt)
      #pragma unroll
      for (int part=0; part<2; ++part){
        int k0 = kt*32 + part*16 + hlf*8;
        float4 f0 = *(const float4*)(xr + k0);
        float4 f1 = *(const float4*)(xr + k0 + 4);
        A[kt].e[part*8+0] = (_Float16)f0.x; A[kt].e[part*8+1] = (_Float16)f0.y;
        A[kt].e[part*8+2] = (_Float16)f0.z; A[kt].e[part*8+3] = (_Float16)f0.w;
        A[kt].e[part*8+4] = (_Float16)f1.x; A[kt].e[part*8+5] = (_Float16)f1.y;
        A[kt].e[part*8+6] = (_Float16)f1.z; A[kt].e[part*8+7] = (_Float16)f1.w;
      }
  }

  v8f acc[4];
  #pragma unroll
  for (int nt=0; nt<4; ++nt){
    float bz = bias ? bias[nt*16 + l16] : 0.0f;
    #pragma unroll
    for (int r=0; r<8; ++r) acc[nt][r] = bz;
  }
  #pragma unroll
  for (int nt=0; nt<4; ++nt)
    #pragma unroll
    for (int kt=0; kt<2; ++kt)
      acc[nt] = __builtin_amdgcn_wmma_f32_16x16x32_f16(
          false, A[kt].v, false, B[nt*2+kt].v, (short)0, acc[nt], false, false);

  if (Y){
    if (fullTile){
      #pragma unroll
      for (int nt=0; nt<4; ++nt)
        #pragma unroll
        for (int r=0; r<8; ++r)
          Y[(size_t)(base + r + 8*hlf)*64 + nt*16 + l16] = acc[nt][r];
    } else {
      #pragma unroll
      for (int nt=0; nt<4; ++nt)
        #pragma unroll
        for (int r=0; r<8; ++r){
          int m = base + r + 8*hlf;
          if (m < N) Y[(size_t)m*64 + nt*16 + l16] = acc[nt][r];
        }
    }
  }
  if (a1) att_reduce_store_(acc, att1, a1, heads, base, hlf, l16, N);
  if (a2) att_reduce_store_(acc, att2, a2, heads, base, hlf, l16, N);
}

// ---------------- segment softmax + scatter aggregation ----------------------
__global__ void init_softmax_kernel(float* __restrict__ amax, float* __restrict__ den, int n){
  int i = blockIdx.x*blockDim.x + threadIdx.x;
  if (i < n){ amax[i] = -3.0e38f; den[i] = 0.0f; }
}

__global__ void edge_max_kernel(const int* __restrict__ src, const int* __restrict__ dst,
                                int E, int nloop, const float* __restrict__ a_s,
                                const float* __restrict__ a_d, float* __restrict__ amax, int heads){
  int i = blockIdx.x*blockDim.x + threadIdx.x;
  if (i >= E + nloop) return;
  int s, d;
  if (i < E){ s = src[i]; d = dst[i]; if (s == d) return; } else { s = d = i - E; }
  for (int h=0; h<heads; ++h){
    float al = lrelu02_(a_s[(size_t)s*heads+h] + a_d[(size_t)d*heads+h]);
    atomicMaxF_(&amax[(size_t)d*heads+h], al);
  }
}

__global__ void edge_sum_kernel(const int* __restrict__ src, const int* __restrict__ dst,
                                int E, int nloop, const float* __restrict__ a_s,
                                const float* __restrict__ a_d, const float* __restrict__ amax,
                                float* __restrict__ den, int heads){
  int i = blockIdx.x*blockDim.x + threadIdx.x;
  if (i >= E + nloop) return;
  int s, d;
  if (i < E){ s = src[i]; d = dst[i]; if (s == d) return; } else { s = d = i - E; }
  for (int h=0; h<heads; ++h){
    float al = lrelu02_(a_s[(size_t)s*heads+h] + a_d[(size_t)d*heads+h]);
    atomicAdd(&den[(size_t)d*heads+h], __expf(al - amax[(size_t)d*heads+h]));
  }
}

__global__ void edge_aggr_kernel(const int* __restrict__ src, const int* __restrict__ dst,
                                 int E, int nloop, const float* __restrict__ a_s,
                                 const float* __restrict__ a_d, const float* __restrict__ amax,
                                 const float* __restrict__ den, const float* __restrict__ feat,
                                 float* __restrict__ out, int heads){
  int i = blockIdx.x*blockDim.x + threadIdx.x;
  int total = (E + nloop)*heads;
  if (i >= total) return;
  int e = i / heads, h = i - e*heads;
  int s, d;
  if (e < E){ s = src[e]; d = dst[e]; if (s == d) return; } else { s = d = e - E; }
  float al = lrelu02_(a_s[(size_t)s*heads+h] + a_d[(size_t)d*heads+h]);
  float w  = __expf(al - amax[(size_t)d*heads+h]) / (den[(size_t)d*heads+h] + 1e-16f);
  int C = 64/heads;
  const float* fs = feat + (size_t)s*64 + h*C;
  float*       op = out  + (size_t)d*64 + h*C;
  for (int c=0; c<C; ++c) atomicAdd(&op[c], w*fs[c]);
}

__global__ void bn_act_kernel(const float* __restrict__ x, const float* __restrict__ g,
                              const float* __restrict__ b, float* __restrict__ y,
                              int total, int relu){
  int i = blockIdx.x*blockDim.x + threadIdx.x;
  if (i >= total) return;
  int col = i & 63;
  float v = x[i]*(g[col]*BN_INV_) + b[col];
  y[i] = relu ? fmaxf(v, 0.0f) : (v > 0.0f ? v : __expf(v) - 1.0f);
}

// ---------------- host-side layer driver -------------------------------------
static void run_gat(const float* xs, const float* xd, bool same,
                    const _Float16* WsT, const _Float16* WdT, const _Float16* WresT,
                    const float* att_s, const float* att_d, const float* bias,
                    const int* src, const int* dst, int E, int nloop, int heads,
                    float* feat, float* AS, float* AD, float* AMX, float* DEN,
                    float* out, int Ns, int Nd, hipStream_t stream)
{
  int gb_s = (Ns + 127)/128, gb_d = (Nd + 127)/128;
  if (same){
    gemm64_wmma_kernel<<<gb_s,256,0,stream>>>(xs, WsT, nullptr, feat, att_s, AS, att_d, AD, heads, Ns);
  } else {
    gemm64_wmma_kernel<<<gb_s,256,0,stream>>>(xs, WsT, nullptr, feat, att_s, AS, nullptr, nullptr, heads, Ns);
    gemm64_wmma_kernel<<<gb_d,256,0,stream>>>(xd, WdT, nullptr, nullptr, att_d, AD, nullptr, nullptr, heads, Nd);
  }
  gemm64_wmma_kernel<<<gb_d,256,0,stream>>>(xd, WresT, bias, out, nullptr, nullptr, nullptr, nullptr, heads, Nd);
  int tot = Nd*heads;
  init_softmax_kernel<<<(tot+255)/256,256,0,stream>>>(AMX, DEN, tot);
  int te = E + nloop;
  edge_max_kernel<<<(te+255)/256,256,0,stream>>>(src, dst, E, nloop, AS, AD, AMX, heads);
  edge_sum_kernel<<<(te+255)/256,256,0,stream>>>(src, dst, E, nloop, AS, AD, AMX, DEN, heads);
  int ta = te*heads;
  edge_aggr_kernel<<<(ta+255)/256,256,0,stream>>>(src, dst, E, nloop, AS, AD, AMX, DEN, feat, out, heads);
}

extern "C" void kernel_launch(void* const* d_in, const int* in_sizes, int n_in,
                              void* d_out, int out_size, void* d_ws, size_t ws_size,
                              hipStream_t stream) {
  (void)in_sizes; (void)n_in; (void)out_size; (void)ws_size;
  const float* agent_hist = (const float*)d_in[0];
  const float* lane_nodes = (const float*)d_in[1];
  const int*   e_aa       = (const int*)d_in[2];
  const int*   e_al       = (const int*)d_in[3];
  const float* a_Wih = (const float*)d_in[4];
  const float* a_Whh = (const float*)d_in[5];
  const float* a_bih = (const float*)d_in[6];
  const float* a_bhh = (const float*)d_in[7];
  const float* l_Wih = (const float*)d_in[8];
  const float* l_Whh = (const float*)d_in[9];
  const float* l_bih = (const float*)d_in[10];
  const float* l_bhh = (const float*)d_in[11];
  const float* aa1_W = (const float*)d_in[12], *aa1_as = (const float*)d_in[13],
             * aa1_ad = (const float*)d_in[14], *aa1_R = (const float*)d_in[15],
             * aa1_b = (const float*)d_in[16];
  const float* aa_bn_g = (const float*)d_in[17], *aa_bn_b = (const float*)d_in[18];
  const float* aa2_W = (const float*)d_in[19], *aa2_as = (const float*)d_in[20],
             * aa2_ad = (const float*)d_in[21], *aa2_R = (const float*)d_in[22],
             * aa2_b = (const float*)d_in[23];
  const float* al1_Ws = (const float*)d_in[24], *al1_Wd = (const float*)d_in[25],
             * al1_as = (const float*)d_in[26], *al1_ad = (const float*)d_in[27],
             * al1_R = (const float*)d_in[28], *al1_b = (const float*)d_in[29];
  const float* al_bn_g = (const float*)d_in[30], *al_bn_b = (const float*)d_in[31];
  const float* al2_W = (const float*)d_in[32], *al2_as = (const float*)d_in[33],
             * al2_ad = (const float*)d_in[34], *al2_R = (const float*)d_in[35],
             * al2_b = (const float*)d_in[36];

  float* OUT0 = (float*)d_out;                       // agent_emb   [NA,64]
  float* OUT1 = OUT0 + (size_t)NA_*64;               // agent_social[NA,64]
  float* OUT2 = OUT1 + (size_t)NA_*64;               // agent_map   [NA,64]
  float* OUT3 = OUT2 + (size_t)NA_*64;               // lane_emb    [NL,64]

  // workspace layout
  char* wsb = (char*)d_ws;
  size_t off = 0;
  auto alloc = [&](size_t bytes)->void*{
    void* p = wsb + off; off = (off + bytes + 255) & ~(size_t)255; return p;
  };
  _Float16* aWhhT = (_Float16*)alloc(256*64*2);
  _Float16* lWhhT = (_Float16*)alloc(256*64*2);
  _Float16* aa1WT = (_Float16*)alloc(64*64*2);
  _Float16* aa1RT = (_Float16*)alloc(64*64*2);
  _Float16* aa2WT = (_Float16*)alloc(64*64*2);
  _Float16* aa2RT = (_Float16*)alloc(64*64*2);
  _Float16* al1ST = (_Float16*)alloc(64*64*2);
  _Float16* al1DT = (_Float16*)alloc(64*64*2);
  _Float16* al1RT = (_Float16*)alloc(64*64*2);
  _Float16* al2WT = (_Float16*)alloc(64*64*2);
  _Float16* al2RT = (_Float16*)alloc(64*64*2);
  float* F0  = (float*)alloc((size_t)NA_*64*4);
  float* F1  = (float*)alloc((size_t)NA_*64*4);
  float* F2  = (float*)alloc((size_t)NA_*64*4);
  float* AS  = (float*)alloc((size_t)NA_*4*4);
  float* AD  = (float*)alloc((size_t)NA_*4*4);
  float* AMX = (float*)alloc((size_t)NA_*4*4);
  float* DEN = (float*)alloc((size_t)NA_*4*4);

  // weight conversions (f32 [K,N] -> f16 [N,K])
  transconv_f16_kernel<<<(256*64+255)/256,256,0,stream>>>(a_Whh, aWhhT, 64, 256);
  transconv_f16_kernel<<<(256*64+255)/256,256,0,stream>>>(l_Whh, lWhhT, 64, 256);
  transconv_f16_kernel<<<16,256,0,stream>>>(aa1_W, aa1WT, 64, 64);
  transconv_f16_kernel<<<16,256,0,stream>>>(aa1_R, aa1RT, 64, 64);
  transconv_f16_kernel<<<16,256,0,stream>>>(aa2_W, aa2WT, 64, 64);
  transconv_f16_kernel<<<16,256,0,stream>>>(aa2_R, aa2RT, 64, 64);
  transconv_f16_kernel<<<16,256,0,stream>>>(al1_Ws, al1ST, 64, 64);
  transconv_f16_kernel<<<16,256,0,stream>>>(al1_Wd, al1DT, 64, 64);
  transconv_f16_kernel<<<16,256,0,stream>>>(al1_R, al1RT, 64, 64);
  transconv_f16_kernel<<<16,256,0,stream>>>(al2_W, al2WT, 64, 64);
  transconv_f16_kernel<<<16,256,0,stream>>>(al2_R, al2RT, 64, 64);

  // LSTMs
  lstm_wmma_kernel<5,false><<<(NA_+127)/128,256,0,stream>>>(
      agent_hist, aWhhT, a_Wih, a_bih, a_bhh, OUT0, nullptr, NA_, TA_);
  lstm_wmma_kernel<2,true><<<(NL_+127)/128,256,0,stream>>>(
      lane_nodes, lWhhT, l_Wih, l_bih, l_bhh, nullptr, OUT3, NL_, TL_);

  const int* src_aa = e_aa;            // row 0
  const int* dst_aa = e_aa + EAA_;     // row 1
  const int* src_al = e_al + EAL_;     // flipped: row 1 = lane (source)
  const int* dst_al = e_al;            // row 0 = agent (dest)

  // aa1: agent_emb -> F1, then BN+ELU -> F2
  run_gat(OUT0, OUT0, true, aa1WT, aa1WT, aa1RT, aa1_as, aa1_ad, aa1_b,
          src_aa, dst_aa, EAA_, NA_, 4, F0, AS, AD, AMX, DEN, F1, NA_, NA_, stream);
  bn_act_kernel<<<((NA_*64)+255)/256,256,0,stream>>>(F1, aa_bn_g, aa_bn_b, F2, NA_*64, 0);

  // aa2: F2 -> agent_social (OUT1)
  run_gat(F2, F2, true, aa2WT, aa2WT, aa2RT, aa2_as, aa2_ad, aa2_b,
          src_aa, dst_aa, EAA_, NA_, 4, F0, AS, AD, AMX, DEN, OUT1, NA_, NA_, stream);

  // al1 (bipartite lane->agent): -> F1, BN+ReLU -> F2
  run_gat(OUT3, OUT1, false, al1ST, al1DT, al1RT, al1_as, al1_ad, al1_b,
          src_al, dst_al, EAL_, NL_, 2, F0, AS, AD, AMX, DEN, F1, NL_, NA_, stream);
  bn_act_kernel<<<((NA_*64)+255)/256,256,0,stream>>>(F1, al_bn_g, al_bn_b, F2, NA_*64, 1);

  // al2 (homogeneous over agents): F2 -> agent_map (OUT2)
  run_gat(F2, F2, true, al2WT, al2WT, al2RT, al2_as, al2_ad, al2_b,
          src_al, dst_al, EAL_, NA_, 2, F0, AS, AD, AMX, DEN, OUT2, NA_, NA_, stream);
}
